// MultiLayer_BTreeLSTM_67190468378861
// MI455X (gfx1250) — matrix operand order repaired and checked by reading, not compiled, not run on hardware
//
#include <hip/hip_runtime.h>
#include <hip/hip_bf16.h>

#if defined(__gfx1250__) && __has_builtin(__builtin_amdgcn_s_wait_asynccnt)
#define ASYNC_LDS_OK 1
#endif

typedef __bf16 bf16_t;
typedef __attribute__((ext_vector_type(8)))  bf16_t v8bf;
typedef __attribute__((ext_vector_type(16))) bf16_t v16bf;
typedef __attribute__((ext_vector_type(8)))  float  v8f;

#define T_LEN 1024
#define NB    64
#define FEATD 512
#define HID   256
#define N5H   1280
#define NTOT  1536   // 5H gates + H highway proj
#define MROWS (T_LEN * NB)

static __device__ __forceinline__ v16bf cat8(v8bf lo, v8bf hi) {
  return __builtin_shufflevector(lo, hi, 0,1,2,3,4,5,6,7,8,9,10,11,12,13,14,15);
}

#ifdef ASYNC_LDS_OK
// GLOBAL_LOAD_ASYNC_TO_LDS_B128 (GV mode): vdst = LDS byte address (low 32
// bits of the generic LDS pointer), vaddr = 64-bit global address, saddr=off.
// Tracked by ASYNCcnt; paired with __builtin_amdgcn_s_wait_asynccnt.
static __device__ __forceinline__ void async_cp16B(const bf16_t* g, bf16_t* l) {
  unsigned int       lofs = (unsigned int)(unsigned long long)(const void*)l;
  unsigned long long ga   = (unsigned long long)(const void*)g;
  asm volatile("global_load_async_to_lds_b128 %0, %1, off"
               :: "v"(lofs), "v"(ga) : "memory");
}
#endif

// copy 32 bytes (16 bf16) global -> LDS
static __device__ __forceinline__ void cp32B(const bf16_t* g, bf16_t* l) {
#ifdef ASYNC_LDS_OK
  async_cp16B(g, l);
  async_cp16B(g + 8, l + 8);
#else
  *(uint4*)l       = *(const uint4*)g;
  *(uint4*)(l + 8) = *(const uint4*)(g + 8);
#endif
}

static __device__ __forceinline__ void wait_async_all() {
#ifdef ASYNC_LDS_OK
  __builtin_amdgcn_s_wait_asynccnt(0);
#endif
}

// ---------------------------------------------------------------------------
// features [B,T,F] f32  ->  x [T,B,F] bf16
// ---------------------------------------------------------------------------
__global__ __launch_bounds__(256) void convert_x_kernel(
    const float* __restrict__ feat, bf16_t* __restrict__ x)
{
  long idx = (long)blockIdx.x * 256 + threadIdx.x;   // over B*T*F
  int  f   = (int)(idx % FEATD);
  long bt  = idx / FEATD;
  int  t   = (int)(bt % T_LEN);
  int  b   = (int)(bt / T_LEN);
  x[((long)t * NB + b) * FEATD + f] = (bf16_t)feat[idx];
}

// ---------------------------------------------------------------------------
// One-time weight prep: Wt[ld][n][k] = bf16( n<1280 ? Wx[ld][k][n]
//                                                   : Wp[ld][k][n-1280] )
// Transposed [N][K] layout makes GEMM B-tiles contiguous 16B async copies.
// ---------------------------------------------------------------------------
__global__ __launch_bounds__(256) void convert_w_kernel(
    const float* __restrict__ Wx, const float* __restrict__ Wp,
    bf16_t* __restrict__ Wt)
{
  long idx = (long)blockIdx.x * 256 + threadIdx.x;   // over 4*1536*512
  int  k   = (int)(idx & (FEATD - 1));
  long nn  = idx >> 9;
  int  n   = (int)(nn % NTOT);
  int  ld  = (int)(nn / NTOT);
  float v = (n < N5H) ? Wx[((long)ld * FEATD + k) * N5H + n]
                      : Wp[((long)ld * FEATD + k) * HID + (n - N5H)];
  Wt[idx] = (bf16_t)v;
}

// ---------------------------------------------------------------------------
// gp = x @ Wt^T + [bx+bh | bp]    (M=65536, K=512, N=1536), bf16 out
// 128x128x32 tiles, double-buffered async-to-LDS staging,
// 8 waves (2M x 4N), wave tile 64x32 = 8 WMMA accumulators
// ---------------------------------------------------------------------------
__global__ __launch_bounds__(256) void gemm_gp_kernel(
    const bf16_t* __restrict__ X,     // [M, 512] bf16
    const bf16_t* __restrict__ Wt,    // [1536, 512] bf16 (this l,d)
    const float*  __restrict__ bxv,
    const float*  __restrict__ bhv,
    const float*  __restrict__ bpv,
    bf16_t* __restrict__ gpout)       // [M, 1536] bf16
{
  __shared__ __align__(16) bf16_t sA[2][128 * 32];   // [m][k] row-major
  __shared__ __align__(16) bf16_t sB[2][128 * 32];   // [n][k] row-major

  const int tid = threadIdx.x;
  const int gm  = blockIdx.x * 128;
  const int gn  = blockIdx.y * 128;

  const int lane = tid & 31, l16 = lane & 15, hi = lane >> 4;
  const int wid  = tid >> 5, wm = wid & 1, wn = wid >> 1;

  v8f acc[4][2];
#pragma unroll
  for (int mt = 0; mt < 4; ++mt)
#pragma unroll
    for (int nt = 0; nt < 2; ++nt) acc[mt][nt] = (v8f)0.0f;

  const int ar = tid >> 1;          // staged row 0..127
  const int ac = (tid & 1) * 16;    // half-offset 0 / 16
  const bf16_t* Asrc = X  + (long)(gm + ar) * FEATD + ac;
  const bf16_t* Bsrc = Wt + (long)(gn + ar) * FEATD + ac;

#define ISSUE_STAGE(kk, buf)                              \
  do {                                                    \
    cp32B(Asrc + (kk) * 32, &sA[buf][ar * 32 + ac]);      \
    cp32B(Bsrc + (kk) * 32, &sB[buf][ar * 32 + ac]);      \
  } while (0)

  ISSUE_STAGE(0, 0);

#pragma unroll 1
  for (int kk = 0; kk < FEATD / 32; ++kk) {
    const int cur = kk & 1;
    if (kk + 1 < FEATD / 32) ISSUE_STAGE(kk + 1, cur ^ 1);
#ifdef ASYNC_LDS_OK
    if (kk + 1 < FEATD / 32) __builtin_amdgcn_s_wait_asynccnt(4);
    else                     __builtin_amdgcn_s_wait_asynccnt(0);
#endif
    __syncthreads();

    v16bf afr[4];
#pragma unroll
    for (int mt = 0; mt < 4; ++mt) {
      const bf16_t* p = &sA[cur][(wm * 64 + mt * 16 + l16) * 32];
      afr[mt] = cat8(*(const v8bf*)(p + hi * 8), *(const v8bf*)(p + 16 + hi * 8));
    }
    v16bf bfr[2];
#pragma unroll
    for (int nt = 0; nt < 2; ++nt) {
      const bf16_t* p = &sB[cur][(wn * 32 + nt * 16 + l16) * 32 + hi * 8];
      bfr[nt] = cat8(*(const v8bf*)p, *(const v8bf*)(p + 16));
    }
#pragma unroll
    for (int mt = 0; mt < 4; ++mt)
#pragma unroll
      for (int nt = 0; nt < 2; ++nt)
        acc[mt][nt] = __builtin_amdgcn_wmma_f32_16x16x32_bf16(
            false, afr[mt], false, bfr[nt], (short)0, acc[mt][nt], false, false);
    __syncthreads();
  }
#undef ISSUE_STAGE

  // ---- epilogue: add fused bias, store bf16
#pragma unroll
  for (int nt = 0; nt < 2; ++nt) {
    int n = gn + wn * 32 + nt * 16 + l16;
    float bias = (n < N5H) ? (bxv[n] + bhv[n]) : bpv[n - N5H];
#pragma unroll
    for (int mt = 0; mt < 4; ++mt) {
#pragma unroll
      for (int v = 0; v < 8; ++v) {
        long m = gm + wm * 64 + mt * 16 + hi * 8 + v;
        gpout[m * NTOT + n] = (bf16_t)(acc[mt][nt][v] + bias);
      }
    }
  }
}

// ---------------------------------------------------------------------------
// Global split-barrier over the 4 persistent workgroups of one direction
// ---------------------------------------------------------------------------
static __device__ __forceinline__ void global_barrier(int* bar, int& target) {
  __syncthreads();
  if (threadIdx.x == 0) {
    __threadfence();
    atomicAdd(bar, 1);
    target += 4;   // gridDim.x == 4
    while (__hip_atomic_load(bar, __ATOMIC_ACQUIRE, __HIP_MEMORY_SCOPE_AGENT) < target)
      __builtin_amdgcn_s_sleep(1);
  }
  __syncthreads();
}

// ---------------------------------------------------------------------------
// Persistent recurrence scan for one direction.
// 4 WGs; WG w owns h-columns [w*64, w*64+64) of all 5 gates + highway col.
// Wh slice (256 x 320, bf16, 160 KB) resident in LDS. Per step:
//   z = h @ Wh_slice with gate-aligned WMMA tiling so all 5 gate
//   pre-activations of a (b,j) land in one lane's registers;
//   gates + cell state stay in VGPRs for all 1024 steps.
// ---------------------------------------------------------------------------
__global__ __launch_bounds__(256) void scan_dir_kernel(
    const bf16_t* __restrict__ gp,    // [T*B, 1536] bf16 (biases folded)
    const float*  __restrict__ Whm,   // [256, 1280] f32
    bf16_t* __restrict__ xnext,       // [T*B, 512]  bf16 (next-layer input)
    float*  __restrict__ dout,        // [B, T, 512] f32 (only if last)
    bf16_t* __restrict__ hbuf,        // [64, 256]   bf16 (cross-WG h)
    int* __restrict__ bar,
    const int dir, const int last)
{
  extern __shared__ __align__(16) char smem[];
  bf16_t* sW = (bf16_t*)smem;              // [320][256]  160 KB
  bf16_t* sH = sW + 320 * 256;             // [64][256]    32 KB

  const int tid  = threadIdx.x;
  const int w    = blockIdx.x;             // slice 0..3
  const int lane = tid & 31, l16 = lane & 15, hi = lane >> 4;
  const int wid  = tid >> 5, wm = wid & 1, wn = wid >> 1;

  // ---- preload Wh slice: sW[g*64 + j][k] = Wh[k][g*256 + w*64 + j]
  {
    const int k = tid;                     // 0..255
    const float* row = Whm + (long)k * N5H + w * 64;
#pragma unroll
    for (int g = 0; g < 5; ++g) {
      const float* rg = row + g * HID;
      for (int j = 0; j < 64; ++j)
        sW[(g * 64 + j) * HID + k] = (bf16_t)rg[j];
    }
  }

  float c[16];
#pragma unroll
  for (int i = 0; i < 16; ++i) c[i] = 0.0f;
  int target = 0;

  const int jcol = wn * 16 + l16;          // 0..63 (local h column)

#pragma unroll 1
  for (int t = 0; t < T_LEN; ++t) {
    const int teff = dir ? (T_LEN - 1 - t) : t;

    // ---- stage h(t-1) into LDS (zero at t==0)
    if (t == 0) {
      uint4 z4 = {0u, 0u, 0u, 0u};
#pragma unroll
      for (int i = 0; i < 8; ++i) *(uint4*)&sH[tid * 64 + i * 8] = z4;
    } else {
#pragma unroll
      for (int i = 0; i < 4; ++i)
        cp32B(hbuf + tid * 64 + i * 16, sH + tid * 64 + i * 16);
      wait_async_all();
    }
    __syncthreads();
    global_barrier(bar, target);           // all WGs consumed h(t-1)

    // ---- z = h @ Wh_slice : wave tile 32(M) x (5 gates x 16 j)
    v8f acc[2][5];
#pragma unroll
    for (int mt = 0; mt < 2; ++mt)
#pragma unroll
      for (int g = 0; g < 5; ++g) acc[mt][g] = (v8f)0.0f;

#pragma unroll 1
    for (int kk = 0; kk < HID / 32; ++kk) {
      const int k0 = kk * 32;
      v16bf afr[2];
#pragma unroll
      for (int mt = 0; mt < 2; ++mt) {
        const bf16_t* p = &sH[(wm * 32 + mt * 16 + l16) * HID + k0];
        afr[mt] = cat8(*(const v8bf*)(p + hi * 8), *(const v8bf*)(p + 16 + hi * 8));
      }
      v16bf bfr[5];
#pragma unroll
      for (int g = 0; g < 5; ++g) {
        const bf16_t* p = &sW[(g * 64 + jcol) * HID + k0 + hi * 8];
        bfr[g] = cat8(*(const v8bf*)p, *(const v8bf*)(p + 16));
      }
#pragma unroll
      for (int mt = 0; mt < 2; ++mt)
#pragma unroll
        for (int g = 0; g < 5; ++g)
          acc[mt][g] = __builtin_amdgcn_wmma_f32_16x16x32_bf16(
              false, afr[mt], false, bfr[g], (short)0, acc[mt][g], false, false);
    }

    // ---- gates + cell state entirely in registers
    const long rowbase = (long)teff * NB;
    const int  hc      = w * 64 + jcol;    // global h column
#pragma unroll
    for (int mt = 0; mt < 2; ++mt) {
#pragma unroll
      for (int v = 0; v < 8; ++v) {
        const int  b    = wm * 32 + mt * 16 + hi * 8 + v;
        const int  ci   = mt * 8 + v;
        const bf16_t* grow = gp + (rowbase + b) * NTOT;
        float zi = acc[mt][0][v] + (float)grow[0 * HID + hc];
        float zo = acc[mt][1][v] + (float)grow[1 * HID + hc];
        float zf = acc[mt][2][v] + (float)grow[2 * HID + hc];
        float zu = acc[mt][3][v] + (float)grow[3 * HID + hc];
        float zr = acc[mt][4][v] + (float)grow[4 * HID + hc];
        float pv = (float)grow[N5H + hc];
        float gi = 1.0f / (1.0f + __expf(-zi));
        float go = 1.0f / (1.0f + __expf(-zo));
        float gf = 1.0f / (1.0f + __expf(-zf));
        float gu = tanhf(zu);
        float gr = 1.0f / (1.0f + __expf(-zr));
        float cv = gi * gu + gf * c[ci];
        c[ci] = cv;
        float hh = go * tanhf(cv);
        float hv = gr * hh + (1.0f - gr) * pv;
        bf16_t hb = (bf16_t)hv;
        hbuf[b * HID + hc] = hb;
        xnext[(rowbase + b) * 512 + dir * HID + hc] = hb;
        if (last) dout[((long)b * T_LEN + teff) * 512 + dir * HID + hc] = hv;
      }
    }
    global_barrier(bar, target);           // h(t) published
  }
}

// ---------------------------------------------------------------------------
extern "C" void kernel_launch(void* const* d_in, const int* in_sizes, int n_in,
                              void* d_out, int out_size, void* d_ws, size_t ws_size,
                              hipStream_t stream) {
  const float* features = (const float*)d_in[0];   // [64,1024,512]
  const float* Wx = (const float*)d_in[1];         // [2,2,512,1280]
  const float* bx = (const float*)d_in[2];         // [2,2,1280]
  const float* Wh = (const float*)d_in[3];         // [2,2,256,1280]
  const float* bh = (const float*)d_in[4];         // [2,2,1280]
  const float* Wp = (const float*)d_in[5];         // [2,2,512,256]
  const float* bp = (const float*)d_in[6];         // [2,2,256]

  char* ws = (char*)d_ws;
  const size_t XBYTES  = (size_t)MROWS * FEATD * sizeof(bf16_t);  // 64 MB
  const size_t GPBYTES = (size_t)MROWS * NTOT  * sizeof(bf16_t);  // 192 MB
  bf16_t* xA    = (bf16_t*)(ws);
  bf16_t* xB    = (bf16_t*)(ws + XBYTES);
  bf16_t* gpbuf = (bf16_t*)(ws + 2 * XBYTES);
  bf16_t* hbuf  = (bf16_t*)(ws + 2 * XBYTES + GPBYTES);
  int*    bar   = (int*)   (ws + 2 * XBYTES + GPBYTES + 65536);
  bf16_t* Wt    = (bf16_t*)(ws + 2 * XBYTES + GPBYTES + 131072); // 6 MB

  convert_x_kernel<<<(NB * T_LEN * FEATD) / 256, 256, 0, stream>>>(features, xA);
  convert_w_kernel<<<(4 * NTOT * FEATD) / 256, 256, 0, stream>>>(Wx, Wp, Wt);

  const size_t SCAN_LDS = (size_t)(320 * 256 + 64 * 256) * sizeof(bf16_t); // 192 KB

  bf16_t* xin = xA;
  bf16_t* xout = xB;
  for (int l = 0; l < 2; ++l) {
    for (int d = 0; d < 2; ++d) {
      const size_t idx = (size_t)(l * 2 + d);
      gemm_gp_kernel<<<dim3(MROWS / 128, NTOT / 128), 256, 0, stream>>>(
          xin,
          Wt + idx * (size_t)NTOT * FEATD,
          bx + idx * N5H, bh + idx * N5H, bp + idx * HID,
          gpbuf);
      (void)hipMemsetAsync(bar, 0, sizeof(int), stream);
      scan_dir_kernel<<<4, 256, SCAN_LDS, stream>>>(
          gpbuf, Wh + idx * (size_t)HID * N5H,
          xout, (float*)d_out, hbuf, bar, d, (l == 1) ? 1 : 0);
    }
    bf16_t* tmp = xin; xin = xout; xout = tmp;
  }
}
